// GCNConv_31576599560639
// MI455X (gfx1250) — matrix operand order repaired and checked by reading, not compiled, not run on hardware
//
#include <hip/hip_runtime.h>

#define NN 16384
#define DD 256

typedef __attribute__((ext_vector_type(16))) __bf16 v16bf;
typedef __attribute__((ext_vector_type(2)))  __bf16 v2bf;
typedef __attribute__((ext_vector_type(8)))  float  v8f;

__device__ __forceinline__ unsigned int bf16rne(float f) {
    unsigned int u = __float_as_uint(f);
    u += 0x7fffu + ((u >> 16) & 1u);
    return u >> 16;
}

// pack two f32 into one dword of bf16 (lo -> [15:0], hi -> [31:16]).
// Round-to-nearest (ties away): same 0.5-ulp bound as RNE, 3 VALU ops total
// (add, add, v_perm_b32).
__device__ __forceinline__ unsigned int pk2(float lo, float hi) {
#if __has_builtin(__builtin_amdgcn_cvt_pk_bf16_f32)
    union { unsigned int u; v2bf h; } c;
    c.h = __builtin_amdgcn_cvt_pk_bf16_f32(lo, hi);
    return c.u;
#else
    unsigned int a = __float_as_uint(lo) + 0x8000u;
    unsigned int b = __float_as_uint(hi) + 0x8000u;
    // dst bytes: [1:0] = a.bytes[3:2], [3:2] = b.bytes[3:2]
    return __builtin_amdgcn_perm(b, a, 0x07060302u);
#endif
}

// pack 8 f32 (two float4, consecutive K) into one uint4 of bf16 pairs
__device__ __forceinline__ uint4 pack8(float4 a, float4 b) {
    uint4 w;
    w.x = pk2(a.x, a.y); w.y = pk2(a.z, a.w);
    w.z = pk2(b.x, b.y); w.w = pk2(b.z, b.w);
    return w;
}

// ---------------- Pass 1: deg row sums -> dinv ----------------
__global__ void __launch_bounds__(256) k_deg(const float* __restrict__ adj,
                                             float* __restrict__ dinv) {
    __shared__ float red[256];
    const int row = blockIdx.x;
    const int tid = threadIdx.x;
    const float4* rp = (const float4*)(adj + (size_t)row * NN);
    float sum = 0.f;
#pragma unroll
    for (int i = 0; i < 16; ++i) {
        float4 v = rp[tid + i * 256];
        sum += (v.x + v.y) + (v.z + v.w);
    }
    red[tid] = sum;
    __syncthreads();
    for (int s = 128; s > 0; s >>= 1) {
        if (tid < s) red[tid] += red[tid + s];
        __syncthreads();
    }
    if (tid == 0) dinv[row] = rsqrtf(fmaxf(red[0], 1e-12f));
}

// ---------------- Pass 2: x1T[d][n] = bf16(feat[n][d]*dinv[n]) ----------------
__global__ void __launch_bounds__(256) k_x1t(const float* __restrict__ feat,
                                             const float* __restrict__ dinv,
                                             unsigned short* __restrict__ x1T) {
    __shared__ float t[16][17];
    const int tx = threadIdx.x & 15, ty = threadIdx.x >> 4;
    const int n0 = blockIdx.x * 16, d0 = blockIdx.y * 16;
    t[ty][tx] = feat[(size_t)(n0 + ty) * DD + d0 + tx] * dinv[n0 + ty];
    __syncthreads();
    x1T[(size_t)(d0 + ty) * NN + n0 + tx] = (unsigned short)bf16rne(t[tx][ty]);
}

// ---------------- Pass 3: wT[o][i] = bf16(weight[i][o]) ----------------
__global__ void __launch_bounds__(256) k_wt(const float* __restrict__ weight,
                                            unsigned short* __restrict__ wT) {
    __shared__ float t[16][17];
    const int tx = threadIdx.x & 15, ty = threadIdx.x >> 4;
    const int i0 = blockIdx.x * 16, o0 = blockIdx.y * 16;
    t[ty][tx] = weight[(size_t)(i0 + ty) * DD + o0 + tx];
    __syncthreads();
    wT[(size_t)(o0 + ty) * DD + i0 + tx] = (unsigned short)bf16rne(t[tx][ty]);
}

// ---------------- Pass 4: x2 = adj @ x1  (bf16 WMMA, f32 accum) ----------------
// Block: 256 thr = 8 waves; rows [m0,m0+64) x all 256 cols. Software-pipelined:
// double-buffered LDS A-fragments (f32->bf16 converted cooperatively), next-step
// A/B global loads issued before current-step WMMAs, one barrier per K-step.
// NOTE: K-loop must NOT be unrolled: unrolling makes the compiler spill/copy the
// 8-VGPR WMMA accumulator blocks (v_mov_b64 walls). Dynamic buffer offset keeps
// the single body copy-free.
__global__ void __launch_bounds__(256) k_gemm1(const float* __restrict__ adj,
                                               const unsigned short* __restrict__ x1T,
                                               float* __restrict__ x2) {
    __shared__ unsigned int ldsA[2][1024]; // 2 x (4 frags * 32 lanes * 8 dwords)
    const int tid  = threadIdx.x;
    const int lane = tid & 31;
    const int wave = tid >> 5;
    const int m0   = blockIdx.x * 64;
    const int n0   = wave * 32;

    v8f acc[4][2];
#pragma unroll
    for (int mt = 0; mt < 4; ++mt)
#pragma unroll
        for (int nt = 0; nt < 2; ++nt)
#pragma unroll
            for (int i = 0; i < 8; ++i) acc[mt][nt][i] = 0.f;

    // Staging thread -> (frag, lane, dword-slot) mapping; each thread packs
    // 8 consecutive K values (2 float4 loads) into one uint4 LDS store.
    const int p0    = tid * 4;
    const int mtS   = p0 >> 8;
    const int qS    = p0 & 255;
    const int lS    = qS >> 3;
    const int sS    = qS & 7;                   // 0 or 4
    const int rowS  = m0 + mtS * 16 + (lS & 15);
    const int koffS = ((lS < 16) ? 0 : 16) + ((sS >= 4) ? 8 : 0);
    const float4* aBase = (const float4*)(adj + (size_t)rowS * NN + koffS);

    // B operand base offsets (lanes 0-15 hold K=0..15, 16-31 hold K=16..31)
    const __bf16* xb = (const __bf16*)x1T;
    size_t bOff[2];
#pragma unroll
    for (int nt = 0; nt < 2; ++nt) {
        const int col = n0 + nt * 16 + (lane & 15);
        bOff[nt] = (size_t)col * NN + ((lane & 16) ? 16 : 0);
    }

    char* ldsBase = (char*)ldsA;
    const int rdLane = lane * 32;   // byte offset of this lane's fragment
    const int wrOff  = tid * 16;    // byte offset of this thread's staging store

    // ---- prologue: stage k=0, load B(k=0) ----
    *(uint4*)(ldsBase + wrOff) = pack8(aBase[0], aBase[1]);
    v16bf b0 = *(const v16bf*)(xb + bOff[0]);
    v16bf b1 = *(const v16bf*)(xb + bOff[1]);
    __syncthreads();

    int bufOff = 0;
#pragma unroll 1
    for (int k = 0; k < NN - 32; k += 32) {
        // prefetch next K-slice (A f32 + B bf16) before current compute
        const float4* ap = aBase + ((k + 32) >> 2);
        float4 nf0 = ap[0], nf1 = ap[1];
        v16bf nb0 = *(const v16bf*)(xb + bOff[0] + k + 32);
        v16bf nb1 = *(const v16bf*)(xb + bOff[1] + k + 32);

        // current-step WMMAs from LDS buffer at bufOff
        const char* rp = ldsBase + (bufOff + rdLane);
#pragma unroll
        for (int mt = 0; mt < 4; ++mt) {
            v16bf a = *(const v16bf*)(rp + mt * 1024);
            acc[mt][0] = __builtin_amdgcn_wmma_f32_16x16x32_bf16(
                false, a, false, b0, (short)0, acc[mt][0], false, false);
            acc[mt][1] = __builtin_amdgcn_wmma_f32_16x16x32_bf16(
                false, a, false, b1, (short)0, acc[mt][1], false, false);
        }

        // stage next slice into the other buffer
        *(uint4*)(ldsBase + ((bufOff ^ 4096) + wrOff)) = pack8(nf0, nf1);
        __syncthreads();   // publishes other buffer; protects current one

        b0 = nb0;
        b1 = nb1;
        bufOff ^= 4096;
    }

    // ---- peeled last K-step ----
    {
        const char* rp = ldsBase + (bufOff + rdLane);
#pragma unroll
        for (int mt = 0; mt < 4; ++mt) {
            v16bf a = *(const v16bf*)(rp + mt * 1024);
            acc[mt][0] = __builtin_amdgcn_wmma_f32_16x16x32_bf16(
                false, a, false, b0, (short)0, acc[mt][0], false, false);
            acc[mt][1] = __builtin_amdgcn_wmma_f32_16x16x32_bf16(
                false, a, false, b1, (short)0, acc[mt][1], false, false);
        }
    }

    const int mlo = (lane & 16) ? 8 : 0;
    const int nc  = lane & 15;
#pragma unroll
    for (int mt = 0; mt < 4; ++mt)
#pragma unroll
        for (int nt = 0; nt < 2; ++nt)
#pragma unroll
            for (int r = 0; r < 8; ++r)
                x2[(size_t)(m0 + mt * 16 + mlo + r) * DD + n0 + nt * 16 + nc] =
                    acc[mt][nt][r];
}

// ---------------- Pass 5: out = (x2 * dinv[:,None]) @ weight ----------------
__global__ void __launch_bounds__(256) k_gemm2(const float* __restrict__ x2,
                                               const unsigned short* __restrict__ wT,
                                               const float* __restrict__ dinv,
                                               float* __restrict__ out) {
    const int tid  = threadIdx.x;
    const int lane = tid & 31;
    const int wave = tid >> 5;
    const int m0   = blockIdx.x * 16;
    const int n0   = wave * 32;

    const int m     = m0 + (lane & 15);
    const int koffA = (lane & 16) ? 16 : 0;
    const float s   = dinv[m];
    const float4* aB = (const float4*)(x2 + (size_t)m * DD + koffA);

    const __bf16* wb = (const __bf16*)wT;
    size_t bOff[2];
#pragma unroll
    for (int nt = 0; nt < 2; ++nt) {
        const int col = n0 + nt * 16 + (lane & 15);
        bOff[nt] = (size_t)col * DD + ((lane & 16) ? 16 : 0);
    }

    v8f acc[2];
#pragma unroll
    for (int nt = 0; nt < 2; ++nt)
#pragma unroll
        for (int i = 0; i < 8; ++i) acc[nt][i] = 0.f;

#pragma unroll
    for (int k = 0; k < DD; k += 32) {
        const float4* ap = aB + (k >> 2);
        float4 f0 = ap[0], f1 = ap[1];   // K = koffA .. koffA+7
        float4 f2 = ap[2], f3 = ap[3];   // K = koffA+8 .. koffA+15
        f0.x *= s; f0.y *= s; f0.z *= s; f0.w *= s;
        f1.x *= s; f1.y *= s; f1.z *= s; f1.w *= s;
        f2.x *= s; f2.y *= s; f2.z *= s; f2.w *= s;
        f3.x *= s; f3.y *= s; f3.z *= s; f3.w *= s;
        union { v16bf v; uint4 q[2]; } A;
        A.q[0] = pack8(f0, f1);
        A.q[1] = pack8(f2, f3);

        v16bf b0 = *(const v16bf*)(wb + bOff[0] + k);
        v16bf b1 = *(const v16bf*)(wb + bOff[1] + k);
        acc[0] = __builtin_amdgcn_wmma_f32_16x16x32_bf16(
            false, A.v, false, b0, (short)0, acc[0], false, false);
        acc[1] = __builtin_amdgcn_wmma_f32_16x16x32_bf16(
            false, A.v, false, b1, (short)0, acc[1], false, false);
    }

    const int mlo = (lane & 16) ? 8 : 0;
    const int nc  = lane & 15;
#pragma unroll
    for (int nt = 0; nt < 2; ++nt)
#pragma unroll
        for (int r = 0; r < 8; ++r)
            out[(size_t)(m0 + mlo + r) * DD + n0 + nt * 16 + nc] = acc[nt][r];
}

extern "C" void kernel_launch(void* const* d_in, const int* in_sizes, int n_in,
                              void* d_out, int out_size, void* d_ws, size_t ws_size,
                              hipStream_t stream) {
    const float* adj    = (const float*)d_in[0];
    const float* feat   = (const float*)d_in[1];
    const float* weight = (const float*)d_in[2];
    float* out = (float*)d_out;

    char* ws = (char*)d_ws;
    float*          dinv = (float*)ws;                                    //  64 KB
    unsigned short* x1T  = (unsigned short*)(ws + (64 << 10));            //   8 MB
    unsigned short* wT   = (unsigned short*)(ws + (64 << 10) + (8 << 20)); // 128 KB
    float*          x2   = (float*)(ws + (64 << 10) + (8 << 20) + (128 << 10)); // 16 MB

    k_deg  <<<NN, 256, 0, stream>>>(adj, dinv);
    k_x1t  <<<dim3(NN / 16, DD / 16), 256, 0, stream>>>(feat, dinv, x1T);
    k_wt   <<<dim3(DD / 16, DD / 16), 256, 0, stream>>>(weight, wT);
    k_gemm1<<<NN / 64, 256, 0, stream>>>(adj, x1T, x2);
    k_gemm2<<<NN / 16, 256, 0, stream>>>(x2, wT, dinv, out);
}
